// GCNModel_39986145525986
// MI455X (gfx1250) — matrix operand order repaired and checked by reading, not compiled, not run on hardware
//
#include <hip/hip_runtime.h>
#include <hip/hip_bf16.h>
#include <cstdint>

// ---------------- problem constants (match reference) ----------------
#define N_NODES   100000
#define N_EDGES   1600000
#define IN_DIM    512
#define HID_DIM   64
#define N_CLASSES 10
#define N_GRAPHS  64
#define C_PAD     16        // pad 10 classes to one 16-wide WMMA tile

typedef __attribute__((ext_vector_type(16))) _Float16 v16h;
typedef __attribute__((ext_vector_type(8)))  _Float16 v8h;
typedef __attribute__((ext_vector_type(8)))  float    v8f;
typedef __attribute__((ext_vector_type(4)))  unsigned int u32x4;
typedef __attribute__((ext_vector_type(8)))  int          i32x8;
typedef __attribute__((ext_vector_type(4)))  int          i32x4;

#if __has_builtin(__builtin_amdgcn_tensor_load_to_lds) && __has_builtin(__builtin_amdgcn_s_wait_tensorcnt)
#define USE_TDM 1
#endif

// ---------------- workspace layout (float elements, 256B-aligned regions) ----
static constexpr size_t OFF_SRCN = 0;                         // 100000 (out_deg -> src_norm)
static constexpr size_t OFF_DSTN = 100032;                    // 100000 (in_deg -> dst_norm)
static constexpr size_t OFF_H1   = 200064;                    // 100000*64  pre-SpMM layer1
static constexpr size_t OFF_AGG1 = OFF_H1   + 6400000;        // 100000*64  SpMM accum (zeroed)
static constexpr size_t OFF_H2   = OFF_AGG1 + 6400000;        // 100000*16  pre-SpMM layer2
static constexpr size_t OFF_AGG2 = OFF_H2   + 1600000;        // 100000*16  SpMM accum (zeroed)
static constexpr size_t OFF_POOL = OFF_AGG2 + 1600000;        // 64*16 graph sums (zeroed)
static constexpr size_t OFF_CNT  = OFF_POOL + (size_t)N_GRAPHS * C_PAD; // 64 counts (zeroed)
static constexpr size_t OFF_W1T  = OFF_CNT  + 64;             // 64*512 halves = 16384 floats (W1^T f16)
static constexpr size_t OFF_W2T  = OFF_W1T  + 16384;          // 16*64 halves = 512 floats   (W2^T padded f16)
static constexpr size_t OFF_H1S  = OFF_W2T  + 512;            // 100000*64 halves = 3.2M floats
static constexpr size_t WS_FLOATS = OFF_H1S + 3200000;        // ~74.1 MiB total

// ---------------- utility kernels ----------------
__global__ void zero_f32(float* __restrict__ p, long n) {
    long i = (long)blockIdx.x * blockDim.x + threadIdx.x;
    long stride = (long)gridDim.x * blockDim.x;
    for (; i < n; i += stride) p[i] = 0.0f;
}

__global__ void degree_k(const int* __restrict__ es, const int* __restrict__ ed,
                         float* __restrict__ outdeg, float* __restrict__ indeg, int ne) {
    int e = blockIdx.x * blockDim.x + threadIdx.x;
    if (e < ne) {
        atomicAdd(&outdeg[es[e]], 1.0f);
        atomicAdd(&indeg[ed[e]], 1.0f);
    }
}

__global__ void rsqrt_k(float* __restrict__ p, int n) {
    int i = blockIdx.x * blockDim.x + threadIdx.x;
    if (i < n) p[i] = rsqrtf(fmaxf(p[i], 1.0f));
}

// W1T[n*512 + k] = (f16) W1[k*64 + n]     (n-major so B-fragments are contiguous)
__global__ void convert_w1t_k(const float* __restrict__ W1, _Float16* __restrict__ w1t) {
    int idx = blockIdx.x * blockDim.x + threadIdx.x;   // 64*512 = 32768
    if (idx < HID_DIM * IN_DIM) {
        int n = idx >> 9;          // /512
        int k = idx & 511;
        w1t[idx] = (_Float16)W1[(size_t)k * HID_DIM + n];
    }
}

// W2T[n*64 + k] = (f16) W2[k*10 + n] for n<10, else 0   (pad classes to 16)
__global__ void convert_w2t_k(const float* __restrict__ W2, _Float16* __restrict__ w2t) {
    int idx = blockIdx.x * blockDim.x + threadIdx.x;   // 16*64 = 1024
    if (idx < C_PAD * HID_DIM) {
        int n = idx >> 6;
        int k = idx & 63;
        float v = (n < N_CLASSES) ? W2[(size_t)k * N_CLASSES + n] : 0.0f;
        w2t[idx] = (_Float16)v;
    }
}

#ifdef USE_TDM
// TDM: DMA the whole 64KB f16 W1^T (32768 x 2B elements, 1-D tile) into LDS.
// D# group0: count=1 | lds_addr | global_addr[56:0] | type=2 ("image")
// D# group1: data_size=1 (2B), tensor_dim0=tile_dim0=32768, stride=32768, dim1=1
__device__ __forceinline__ void tdm_load_w1(const _Float16* w1t, void* smem) {
    unsigned lds_off = (unsigned)(uintptr_t)smem;          // low 32 bits = LDS byte offset
    unsigned long long ga = (unsigned long long)(uintptr_t)w1t;
    u32x4 g0;
    g0[0] = 1u;                                            // count=1, user descriptor
    g0[1] = lds_off;                                       // lds_addr
    g0[2] = (unsigned)(ga & 0xFFFFFFFFu);                  // global_addr[31:0]
    g0[3] = (unsigned)((ga >> 32) & 0x01FFFFFFu) | (2u << 30); // global_addr[56:32] | type=2
    i32x8 g1;
    g1[0] = (int)(1u << 16);          // workgroup_mask=0, data_size=1 (2 bytes)
    g1[1] = (int)(32768u << 16);      // abar_addr=0 | tensor_dim0[15:0]=32768
    g1[2] = (int)(1u << 16);          // tensor_dim0[31:16]=0 | tensor_dim1=1
    g1[3] = (int)(32768u << 16);      // tensor_dim1 hi=0 | tile_dim0=32768
    g1[4] = 0;                        // tile_dim1=0 (unused) | tile_dim2=0
    g1[5] = 32768;                    // tensor_dim0_stride[31:0]
    g1[6] = 0;                        // stride hi | tensor_dim1_stride lo
    g1[7] = 0;
    i32x4 z4 = {0, 0, 0, 0};
#if defined(__clang_major__) && (__clang_major__ >= 23)
    i32x8 z8 = {0, 0, 0, 0, 0, 0, 0, 0};
    __builtin_amdgcn_tensor_load_to_lds(g0, g1, z4, z4, z8, 0);
#else
    __builtin_amdgcn_tensor_load_to_lds(g0, g1, z4, z4, 0);
#endif
}
#endif

// ---------------- layer-1 GEMM: h1[100000x64] = (x .* src_norm) @ W1 ----------
// 1 wave = one 16-row M-tile across ALL 4 N-tiles (A-fragment reused by 4 WMMAs,
// cutting the dominant x-read traffic 4x vs tile-per-wave). 8 waves/block.
__global__ __launch_bounds__(256) void gemm1_wmma(
    const float* __restrict__ x, const float* __restrict__ srcn,
    const _Float16* __restrict__ w1t, float* __restrict__ h1)
{
    __shared__ __align__(32) _Float16 sW[HID_DIM * IN_DIM];   // 64 KB: full W1^T in f16
#ifdef USE_TDM
    if (threadIdx.x == 0) tdm_load_w1(w1t, sW);               // wave 0 issues one TDM op
    __builtin_amdgcn_s_wait_tensorcnt(0);                     // no-op for non-issuing waves
    __syncthreads();
#else
    {
        const uint4* g = reinterpret_cast<const uint4*>(w1t);
        uint4* s = reinterpret_cast<uint4*>(sW);
        #pragma unroll
        for (int i = 0; i < 16; ++i)
            s[threadIdx.x + 256 * i] = g[threadIdx.x + 256 * i];
    }
    __syncthreads();
#endif

    const int wid  = threadIdx.x >> 5;
    const int lane = threadIdx.x & 31;
    const int mtile = blockIdx.x * 8 + wid;
    if (mtile >= N_NODES / 16) return;                        // after the barrier
    const int l16  = lane & 15;
    const bool hi  = lane >= 16;

    const int row = mtile * 16 + l16;
    const float sn = srcn[row];
    const float* xrow = x + (size_t)row * IN_DIM;
    const _Float16* b0p = &sW[(size_t)(l16)      * IN_DIM];
    const _Float16* b1p = &sW[(size_t)(16 + l16) * IN_DIM];
    const _Float16* b2p = &sW[(size_t)(32 + l16) * IN_DIM];
    const _Float16* b3p = &sW[(size_t)(48 + l16) * IN_DIM];

    v8f acc0 = {}, acc1 = {}, acc2 = {}, acc3 = {};
    #pragma unroll 2
    for (int kk = 0; kk < IN_DIM / 32; ++kk) {
        const int k0 = kk * 32 + (hi ? 8 : 0);
        __builtin_prefetch(xrow + k0 + 32, 0, 3);             // next K-chunk (speculative)
        float4 f0 = *reinterpret_cast<const float4*>(xrow + k0);
        float4 f1 = *reinterpret_cast<const float4*>(xrow + k0 + 4);
        float4 f2 = *reinterpret_cast<const float4*>(xrow + k0 + 16);
        float4 f3 = *reinterpret_cast<const float4*>(xrow + k0 + 20);
        v16h a;
        a[0]  = (_Float16)(f0.x * sn); a[1]  = (_Float16)(f0.y * sn);
        a[2]  = (_Float16)(f0.z * sn); a[3]  = (_Float16)(f0.w * sn);
        a[4]  = (_Float16)(f1.x * sn); a[5]  = (_Float16)(f1.y * sn);
        a[6]  = (_Float16)(f1.z * sn); a[7]  = (_Float16)(f1.w * sn);
        a[8]  = (_Float16)(f2.x * sn); a[9]  = (_Float16)(f2.y * sn);
        a[10] = (_Float16)(f2.z * sn); a[11] = (_Float16)(f2.w * sn);
        a[12] = (_Float16)(f3.x * sn); a[13] = (_Float16)(f3.y * sn);
        a[14] = (_Float16)(f3.z * sn); a[15] = (_Float16)(f3.w * sn);

        const int kb = kk * 32 + (hi ? 8 : 0);
        v16h b0 = __builtin_shufflevector(*reinterpret_cast<const v8h*>(b0p + kb), *reinterpret_cast<const v8h*>(b0p + kb + 16), 0,1,2,3,4,5,6,7,8,9,10,11,12,13,14,15);   // 2x16B LDS reads: k = kb..kb+7, kb+16..kb+23
        v16h b1 = __builtin_shufflevector(*reinterpret_cast<const v8h*>(b1p + kb), *reinterpret_cast<const v8h*>(b1p + kb + 16), 0,1,2,3,4,5,6,7,8,9,10,11,12,13,14,15);
        v16h b2 = __builtin_shufflevector(*reinterpret_cast<const v8h*>(b2p + kb), *reinterpret_cast<const v8h*>(b2p + kb + 16), 0,1,2,3,4,5,6,7,8,9,10,11,12,13,14,15);
        v16h b3 = __builtin_shufflevector(*reinterpret_cast<const v8h*>(b3p + kb), *reinterpret_cast<const v8h*>(b3p + kb + 16), 0,1,2,3,4,5,6,7,8,9,10,11,12,13,14,15);

        acc0 = __builtin_amdgcn_wmma_f32_16x16x32_f16(false, a, false, b0, (short)0, acc0, false, false);
        acc1 = __builtin_amdgcn_wmma_f32_16x16x32_f16(false, a, false, b1, (short)0, acc1, false, false);
        acc2 = __builtin_amdgcn_wmma_f32_16x16x32_f16(false, a, false, b2, (short)0, acc2, false, false);
        acc3 = __builtin_amdgcn_wmma_f32_16x16x32_f16(false, a, false, b3, (short)0, acc3, false, false);
    }

    const int mbase = mtile * 16 + (hi ? 8 : 0);
    #pragma unroll
    for (int r = 0; r < 8; ++r) {
        float* orow = h1 + (size_t)(mbase + r) * HID_DIM + l16;
        orow[0]  = acc0[r];
        orow[16] = acc1[r];
        orow[32] = acc2[r];
        orow[48] = acc3[r];
    }
}

// ---------------- SpMM: agg[dst] += h[src]  (F = 64 or 16) ----------------
template<int F>
__global__ void spmm_k(const float* __restrict__ h, const int* __restrict__ es,
                       const int* __restrict__ ed, float* __restrict__ agg, long total) {
    long idx = (long)blockIdx.x * blockDim.x + threadIdx.x;
    if (idx >= total) return;
    int e = (int)(idx / F);
    int f = (int)(idx & (F - 1));
    float v = h[(size_t)es[e] * F + f];
    atomicAdd(&agg[(size_t)ed[e] * F + f], v);
}

// h1s = (f16)( relu(agg1 * dst_norm + b1) * src_norm )  -- src_norm fused for layer-2 A
__global__ void bias_relu_scale_k(const float* __restrict__ agg1,
                                  const float* __restrict__ dstn,
                                  const float* __restrict__ srcn,
                                  const float* __restrict__ b1,
                                  _Float16* __restrict__ h1s) {
    long idx = (long)blockIdx.x * blockDim.x + threadIdx.x;
    if (idx >= (long)N_NODES * HID_DIM) return;
    int row = (int)(idx >> 6);
    int f   = (int)(idx & 63);
    float v = agg1[idx] * dstn[row] + b1[f];
    v = fmaxf(v, 0.0f) * srcn[row];
    h1s[idx] = (_Float16)v;
}

// ---------------- layer-2 GEMM: h2[100000x16] = h1s @ W2pad  (WMMA, K=64) -----
__global__ __launch_bounds__(256) void gemm2_wmma(
    const _Float16* __restrict__ h1s, const _Float16* __restrict__ w2t,
    float* __restrict__ h2)
{
    const int wid  = threadIdx.x >> 5;
    const int lane = threadIdx.x & 31;
    const int mtile = blockIdx.x * 8 + wid;
    if (mtile >= N_NODES / 16) return;
    const int l16 = lane & 15;
    const bool hi = lane >= 16;

    const int row = mtile * 16 + l16;
    const _Float16* arow = h1s + (size_t)row * HID_DIM;

    v8f acc = {};
    #pragma unroll
    for (int kk = 0; kk < HID_DIM / 32; ++kk) {
        const int k0 = kk * 32 + (hi ? 8 : 0);
        v8h g0 = *reinterpret_cast<const v8h*>(arow + k0);
        v8h g1 = *reinterpret_cast<const v8h*>(arow + k0 + 16);
        v16h a;
        #pragma unroll
        for (int j = 0; j < 8; ++j) { a[j] = g0[j]; a[8 + j] = g1[j]; }
        const int kb = kk * 32 + (hi ? 8 : 0);
        v16h b = __builtin_shufflevector(*reinterpret_cast<const v8h*>(w2t + (size_t)l16 * HID_DIM + kb), *reinterpret_cast<const v8h*>(w2t + (size_t)l16 * HID_DIM + kb + 16), 0,1,2,3,4,5,6,7,8,9,10,11,12,13,14,15);
        acc = __builtin_amdgcn_wmma_f32_16x16x32_f16(
                  false, a, false, b, (short)0, acc, false, false);
    }

    const int mbase = mtile * 16 + (hi ? 8 : 0);
    #pragma unroll
    for (int r = 0; r < 8; ++r)
        h2[(size_t)(mbase + r) * C_PAD + l16] = acc[r];
}

// per-node final value -> per-graph sums + counts
__global__ void pool_k(const float* __restrict__ agg2, const float* __restrict__ dstn,
                       const float* __restrict__ b2, const int* __restrict__ n2g,
                       float* __restrict__ pool, float* __restrict__ cnt) {
    long idx = (long)blockIdx.x * blockDim.x + threadIdx.x;
    if (idx >= (long)N_NODES * C_PAD) return;
    int node = (int)(idx >> 4);
    int f    = (int)(idx & 15);
    int g = n2g[node];
    if (f < N_CLASSES) {
        float v = agg2[idx] * dstn[node] + b2[f];
        atomicAdd(&pool[(size_t)g * C_PAD + f], v);
    } else if (f == 15) {
        atomicAdd(&cnt[g], 1.0f);
    }
}

__global__ void finalize_k(const float* __restrict__ pool, const float* __restrict__ cnt,
                           float* __restrict__ out) {
    int idx = blockIdx.x * blockDim.x + threadIdx.x;
    if (idx >= N_GRAPHS * N_CLASSES) return;
    int g = idx / N_CLASSES;
    int f = idx - g * N_CLASSES;
    out[idx] = pool[(size_t)g * C_PAD + f] / fmaxf(cnt[g], 1.0f);
}

// ---------------- host-side launch ----------------
extern "C" void kernel_launch(void* const* d_in, const int* in_sizes, int n_in,
                              void* d_out, int out_size, void* d_ws, size_t ws_size,
                              hipStream_t stream) {
    const float* x   = (const float*)d_in[0];
    const int*   es  = (const int*)d_in[1];
    const int*   ed  = (const int*)d_in[2];
    const int*   n2g = (const int*)d_in[3];
    // d_in[4] = n_graphs (constant 64, matches N_GRAPHS)
    const float* W1  = (const float*)d_in[5];
    const float* b1  = (const float*)d_in[6];
    const float* W2  = (const float*)d_in[7];
    const float* b2  = (const float*)d_in[8];
    float* out = (float*)d_out;

    float* ws = (float*)d_ws;
    float*     srcn = ws + OFF_SRCN;
    float*     dstn = ws + OFF_DSTN;
    float*     h1   = ws + OFF_H1;
    float*     agg1 = ws + OFF_AGG1;
    float*     h2   = ws + OFF_H2;
    float*     agg2 = ws + OFF_AGG2;
    float*     pool = ws + OFF_POOL;
    float*     cnt  = ws + OFF_CNT;
    _Float16*  w1t  = reinterpret_cast<_Float16*>(ws + OFF_W1T);
    _Float16*  w2t  = reinterpret_cast<_Float16*>(ws + OFF_W2T);
    _Float16*  h1s  = reinterpret_cast<_Float16*>(ws + OFF_H1S);

    const int ne = in_sizes[1];   // 1,600,000

    // zero everything that is accumulated into (ws is poisoned, not re-zeroed by harness)
    zero_f32<<<2048, 256, 0, stream>>>(ws, (long)OFF_H1);                          // degree bufs
    zero_f32<<<2048, 256, 0, stream>>>(agg1, (long)N_NODES * HID_DIM);             // layer-1 SpMM
    zero_f32<<<2048, 256, 0, stream>>>(agg2, (long)(OFF_W1T - OFF_AGG2));          // agg2+pool+cnt

    // degrees -> norms (in place)
    degree_k<<<(ne + 255) / 256, 256, 0, stream>>>(es, ed, srcn, dstn, ne);
    rsqrt_k<<<(int)((OFF_H1 + 255) / 256), 256, 0, stream>>>(ws, (int)OFF_H1);

    // f16 transposed weights
    convert_w1t_k<<<(HID_DIM * IN_DIM + 255) / 256, 256, 0, stream>>>(W1, w1t);
    convert_w2t_k<<<(C_PAD * HID_DIM + 255) / 256, 256, 0, stream>>>(W2, w2t);

    // layer 1
    gemm1_wmma<<<(N_NODES / 16 + 7) / 8, 256, 0, stream>>>(x, srcn, w1t, h1);      // 782 blocks
    {
        long total = (long)ne * HID_DIM;                                           // 102.4M
        spmm_k<HID_DIM><<<(int)((total + 255) / 256), 256, 0, stream>>>(h1, es, ed, agg1, total);
    }
    {
        long total = (long)N_NODES * HID_DIM;
        bias_relu_scale_k<<<(int)((total + 255) / 256), 256, 0, stream>>>(agg1, dstn, srcn, b1, h1s);
    }

    // layer 2
    gemm2_wmma<<<(N_NODES / 16 + 7) / 8, 256, 0, stream>>>(h1s, w2t, h2);          // 782 blocks
    {
        long total = (long)ne * C_PAD;                                             // 25.6M
        spmm_k<C_PAD><<<(int)((total + 255) / 256), 256, 0, stream>>>(h2, es, ed, agg2, total);
    }

    // pooling + mean
    {
        long total = (long)N_NODES * C_PAD;
        pool_k<<<(int)((total + 255) / 256), 256, 0, stream>>>(agg2, dstn, b2, n2g, pool, cnt);
    }
    finalize_k<<<(N_GRAPHS * N_CLASSES + 255) / 256, 256, 0, stream>>>(pool, cnt, out);
}
